// EventTransformer_60112362275157
// MI455X (gfx1250) — matrix operand order repaired and checked by reading, not compiled, not run on hardware
//
#include <hip/hip_runtime.h>
#include <math.h>

#define NDOM    65536
#define BATCH   128
#define DIN     128
#define DMODEL  256
#define NHEADS  4
#define HDIM    64
#define NLAYERS 4
#define DFF     1024
#define SMAX    768
#define ROWS    (BATCH * SMAX)   // 98304
#define QBLKS   (SMAX / 128)     // 6 query blocks of 128 per (b,h)
#define KC      64               // key chunk for flash attention
#define NEGBIG  (-1.0e30f)
#define NBANDS  42

// gfx1250 async global->LDS path (ASYNCcnt-tracked, bypasses VGPRs).
// Guarded so the file compiles on toolchains without the builtins.
#if defined(__AMDGCN__) && defined(__has_builtin)
#if __has_builtin(__builtin_amdgcn_global_load_async_to_lds_b128) && \
    __has_builtin(__builtin_amdgcn_s_wait_asynccnt)
#define USE_ASYNC_LDS 1
#endif
#endif
#ifndef USE_ASYNC_LDS
#define USE_ASYNC_LDS 0
#endif

#if USE_ASYNC_LDS
typedef int v4i_ __attribute__((ext_vector_type(4)));
// builtin signature (from hipcc diagnostics): (v4i AS1* gsrc, v4i AS3* ldst, imm, imm)
#define ASYNC_GPTR(p) ((__attribute__((address_space(1))) v4i_*)(p))
#define ASYNC_LPTR(p) ((__attribute__((address_space(3))) v4i_*)(p))
#endif

typedef __attribute__((ext_vector_type(16))) _Float16 f16x16;
typedef __attribute__((ext_vector_type(8)))  float    f32x8;

union FragAB {
  f16x16  v;
  float4  f4[2];
  _Float16 h[16];
};

__device__ __forceinline__ f32x8 wmma16(const FragAB& a, const FragAB& b, f32x8 c) {
  // D(16x16,f32) = A(16x32,f16) * B(32x16,f16) + C
  return __builtin_amdgcn_wmma_f32_16x16x32_f16(
      /*neg_a=*/false, a.v, /*neg_b=*/false, b.v,
      /*c_mod=*/(short)0, c, /*reuse_a=*/false, /*reuse_b=*/false);
}

// A-fragment: lane holds row M=lane&15; K = k0 + (hi?8:0) + {0..7} and +16..+23
__device__ __forceinline__ FragAB load_a_frag(const _Float16* row, int k0, int hi) {
  FragAB f;
  int ka = k0 + (hi ? 8 : 0);
  f.f4[0] = *(const float4*)(row + ka);
  f.f4[1] = *(const float4*)(row + ka + 16);
  return f;
}
// B-fragment: lane holds column N=lane&15; K = k0 + (hi?8:0) + {0..7} and +16..+23
__device__ __forceinline__ FragAB load_b_frag(const _Float16* col, int k0, int hi) {
  FragAB f;
  int kb = k0 + (hi ? 8 : 0);
  f.f4[0] = *(const float4*)(col + kb);
  f.f4[1] = *(const float4*)(col + kb + 16);
  return f;
}

// ---------------------------------------------------------------- helpers

__global__ __launch_bounds__(256) void cvt_kernel(const float* __restrict__ s,
                                                  _Float16* __restrict__ d, int n) {
  int t = blockIdx.x * 256 + threadIdx.x;
  if (t < n) d[t] = (_Float16)s[t];
}

// per-DOM fp16 embeddings + sin/cos positional encoding (252 used, 4 zero pad)
__global__ __launch_bounds__(256) void prep_kernel(const float* __restrict__ emb,
                                                   const float* __restrict__ geom,
                                                   _Float16* __restrict__ embh,
                                                   _Float16* __restrict__ ench) {
  size_t t = (size_t)blockIdx.x * 256 + threadIdx.x;   // over NDOM*256
  int i = (int)(t >> 8), j = (int)(t & 255);
  if (j < DIN) embh[(size_t)i * DIN + j] = (_Float16)emb[(size_t)i * DIN + j];
  float e = 0.f;
  if (j < 3 * NBANDS * 2) {
    int a = j / (NBANDS * 2);
    int r = j % (NBANDS * 2);
    int f = r >> 1, sc = r & 1;
    float freq = __powf(10.f, (float)f / (float)(NBANDS - 1));
    float ang = 6.28318530717958647692f * geom[(size_t)i * 3 + a] * freq;
    e = sc ? __cosf(ang) : __sinf(ang);
  }
  ench[t] = (_Float16)e;
}

// starts[b] = lower_bound over sorted dom_to_event_idx
__global__ void starts_kernel(const int* __restrict__ idx, int* __restrict__ starts) {
  int b = threadIdx.x;                                  // 128 threads
  int lo = 0, hi = NDOM;
  while (lo < hi) { int mid = (lo + hi) >> 1; if (idx[mid] < b) lo = mid + 1; else hi = mid; }
  starts[b] = lo;
}

// h = 0 everywhere except CLS row; mask bias = -1e30 everywhere (CLS key masked too)
__global__ __launch_bounds__(256) void init_kernel(const float* __restrict__ cls,
                                                   float* __restrict__ h,
                                                   float* __restrict__ maskb) {
  size_t t = (size_t)blockIdx.x * 256 + threadIdx.x;    // ROWS*256 threads
  size_t row = t >> 8;
  int d = (int)(t & 255);
  int s = (int)(row % SMAX);
  h[t] = (s == 0) ? cls[d] : 0.f;
  if (d == 0) maskb[row] = NEGBIG;
}

// x = emb@Win^T + enc@Wgeo^T + biases, scattered to h[event][pos]; unmask position
__global__ __launch_bounds__(256) void embed_kernel(
    const _Float16* __restrict__ embh, const _Float16* __restrict__ ench,
    const _Float16* __restrict__ winh, const _Float16* __restrict__ wgeoh,
    const float* __restrict__ bin, const float* __restrict__ bgeo,
    const int* __restrict__ evt, const int* __restrict__ starts,
    float* __restrict__ h, float* __restrict__ maskb) {
  int wave = threadIdx.x >> 5, lane = threadIdx.x & 31;
  int hi = lane >> 4, ln16 = lane & 15;
  int m0 = (blockIdx.x * 8 + wave) * 16;
  int n0 = blockIdx.y * 16;
  const _Float16* a1 = embh + (size_t)(m0 + ln16) * DIN;
  const _Float16* w1 = winh + (size_t)(n0 + ln16) * DIN;
  const _Float16* a2 = ench + (size_t)(m0 + ln16) * DMODEL;
  const _Float16* w2 = wgeoh + (size_t)(n0 + ln16) * DMODEL;
  f32x8 acc = {0.f, 0.f, 0.f, 0.f, 0.f, 0.f, 0.f, 0.f};
  for (int k = 0; k < DIN; k += 32)
    acc = wmma16(load_a_frag(a1, k, hi), load_b_frag(w1, k, hi), acc);
  for (int k = 0; k < DMODEL; k += 32)
    acc = wmma16(load_a_frag(a2, k, hi), load_b_frag(w2, k, hi), acc);
  float bn = bin[n0 + ln16] + bgeo[n0 + ln16];
#pragma unroll
  for (int r = 0; r < 8; ++r) {
    int i = m0 + r + (hi ? 8 : 0);
    int bev = evt[i];
    int p = i - starts[bev] + 1;
    if (p < SMAX) {
      h[((size_t)bev * SMAX + p) * DMODEL + n0 + ln16] = acc[r] + bn;
      if (n0 == 0 && ln16 == 0) maskb[(size_t)bev * SMAX + p] = 0.f;
    }
  }
}

// fp32 LayerNorm, fp16 output; one wave per 256-wide row
__global__ __launch_bounds__(256) void ln_kernel(const float* __restrict__ x,
                                                 const float* __restrict__ w,
                                                 const float* __restrict__ b,
                                                 _Float16* __restrict__ out) {
  int wave = threadIdx.x >> 5, lane = threadIdx.x & 31;
  size_t row = (size_t)blockIdx.x * 8 + wave;
  const float* xr = x + row * DMODEL;
  float v[8];
  float s = 0.f;
#pragma unroll
  for (int k = 0; k < 8; ++k) { v[k] = xr[lane + 32 * k]; s += v[k]; }
#pragma unroll
  for (int m = 16; m >= 1; m >>= 1) s += __shfl_xor(s, m, 32);
  float mean = s * (1.f / DMODEL);
  float q = 0.f;
#pragma unroll
  for (int k = 0; k < 8; ++k) { float d = v[k] - mean; q += d * d; }
#pragma unroll
  for (int m = 16; m >= 1; m >>= 1) q += __shfl_xor(q, m, 32);
  float rstd = rsqrtf(q * (1.f / DMODEL) + 1e-5f);
#pragma unroll
  for (int k = 0; k < 8; ++k) {
    int d = lane + 32 * k;
    out[row * DMODEL + d] = (_Float16)((v[k] - mean) * rstd * w[d] + b[d]);
  }
}

// generic C = act(A@W^T + bias) [+ resid], A: MxK fp16, W: NxK fp16 (row-major)
// one wave per 16x16 C tile; 8 waves share the N-tile (shared B hits L0)
__global__ __launch_bounds__(256) void gemm_kernel(
    const _Float16* __restrict__ A, const _Float16* __restrict__ W,
    const float* __restrict__ bias, const float* resid,
    float* outF, _Float16* outH, int M, int N, int K, int relu) {
  int wave = threadIdx.x >> 5, lane = threadIdx.x & 31;
  int hi = lane >> 4, ln16 = lane & 15;
  int m0 = (blockIdx.x * 8 + wave) * 16;
  int n0 = blockIdx.y * 16;
  if (m0 >= M) return;
  const _Float16* arow = A + (size_t)(m0 + ln16) * K;
  const _Float16* wrow = W + (size_t)(n0 + ln16) * K;
  f32x8 acc = {0.f, 0.f, 0.f, 0.f, 0.f, 0.f, 0.f, 0.f};
  for (int k = 0; k < K; k += 32)
    acc = wmma16(load_a_frag(arow, k, hi), load_b_frag(wrow, k, hi), acc);
  float bn = bias ? bias[n0 + ln16] : 0.f;
#pragma unroll
  for (int r = 0; r < 8; ++r) {
    float v = acc[r] + bn;
    if (relu) v = v > 0.f ? v : 0.f;
    int m = m0 + r + (hi ? 8 : 0);
    size_t off = (size_t)m * N + n0 + ln16;
    if (resid) v += resid[off];
    if (outF) outF[off] = v;
    else      outH[off] = (_Float16)v;
  }
}

// flash attention: block = 8 waves, each wave owns a 16-query tile of one (b,head);
// K chunk staged to LDS via async global->LDS (ASYNCcnt) when available,
// V chunk transposed into LDS, P re-laid-out through wave-private LDS.
__global__ __launch_bounds__(256) void attn_kernel(
    const _Float16* __restrict__ qkv,     // (B*SMAX, 768)
    const float* __restrict__ maskb,      // (B, SMAX) additive bias
    _Float16* __restrict__ attnout) {     // (B*SMAX, 256)
  __shared__ alignas(16) _Float16 Kc[KC][HDIM];       // 8 KB
  __shared__ alignas(16) _Float16 Vt[HDIM][KC];       // 8 KB (d-major)
  __shared__ alignas(16) _Float16 Pb[8][16][KC];      // 16 KB, per-wave scratch

  int wave = threadIdx.x >> 5, lane = threadIdx.x & 31;
  int hi = lane >> 4, ln16 = lane & 15;
  int blk = blockIdx.x;
  int qb = blk % QBLKS;
  int bh = blk / QBLKS;
  int b  = bh >> 2;                 // / NHEADS
  int hh = bh & (NHEADS - 1);
  int q0 = qb * 128 + wave * 16;
  size_t rowbase = (size_t)b * SMAX;

  FragAB qf[2];
  {
    const _Float16* qrow = qkv + (rowbase + q0 + ln16) * 768 + hh * HDIM;
    qf[0] = load_a_frag(qrow, 0, hi);
    qf[1] = load_a_frag(qrow, 32, hi);
  }

  float mrun[8], lrun[8];
  f32x8 oacc[4];
#pragma unroll
  for (int r = 0; r < 8; ++r) { mrun[r] = NEGBIG; lrun[r] = 0.f; }
#pragma unroll
  for (int t = 0; t < 4; ++t) oacc[t] = (f32x8){0.f,0.f,0.f,0.f,0.f,0.f,0.f,0.f};

  // staging coordinates: 4 threads per key row, 16 fp16 each
  int srow = threadIdx.x >> 2, sseg = (threadIdx.x & 3) * 16;

  for (int kc = 0; kc < SMAX; kc += KC) {
    {
      const _Float16* ks = qkv + (rowbase + kc + srow) * 768 + 256 + hh * HDIM + sseg;
      const _Float16* vs = qkv + (rowbase + kc + srow) * 768 + 512 + hh * HDIM + sseg;
      // prefetch next chunk's V rows into cache while this chunk computes
      if (kc + KC < SMAX)
        __builtin_prefetch((const void*)(vs + (size_t)KC * 768), 0, 1);
#if USE_ASYNC_LDS
      // async DMA straight into LDS, tracked on ASYNCcnt (no VGPR round-trip)
      __builtin_amdgcn_global_load_async_to_lds_b128(
          ASYNC_GPTR(ks), ASYNC_LPTR(&Kc[srow][sseg]), 0, 0);
      __builtin_amdgcn_global_load_async_to_lds_b128(
          ASYNC_GPTR(ks + 8), ASYNC_LPTR(&Kc[srow][sseg + 8]), 0, 0);
#else
      *(float4*)(&Kc[srow][sseg])     = *(const float4*)(ks);
      *(float4*)(&Kc[srow][sseg + 8]) = *(const float4*)(ks + 8);
#endif
#pragma unroll
      for (int c = 0; c < 16; ++c) Vt[sseg + c][srow] = vs[c];
#if USE_ASYNC_LDS
      __builtin_amdgcn_s_wait_asynccnt(0);   // wave's async writes landed in LDS
#endif
    }
    __syncthreads();

    // scores S = Q K^T / 8 + mask
    f32x8 sacc[4];
#pragma unroll
    for (int t = 0; t < 4; ++t) sacc[t] = (f32x8){0.f,0.f,0.f,0.f,0.f,0.f,0.f,0.f};
#pragma unroll
    for (int t = 0; t < 4; ++t)
#pragma unroll
      for (int s = 0; s < 2; ++s)
        sacc[t] = wmma16(qf[s], load_b_frag(&Kc[t * 16 + ln16][0], s * 32, hi), sacc[t]);

    float mb[4];
#pragma unroll
    for (int t = 0; t < 4; ++t) mb[t] = maskb[rowbase + kc + t * 16 + ln16];
#pragma unroll
    for (int t = 0; t < 4; ++t)
#pragma unroll
      for (int r = 0; r < 8; ++r) sacc[t][r] = sacc[t][r] * 0.125f + mb[t];

    // online softmax update (per row r+8*hi)
    float nm[8];
#pragma unroll
    for (int r = 0; r < 8; ++r)
      nm[r] = fmaxf(fmaxf(sacc[0][r], sacc[1][r]), fmaxf(sacc[2][r], sacc[3][r]));
#pragma unroll
    for (int m = 8; m >= 1; m >>= 1)
#pragma unroll
      for (int r = 0; r < 8; ++r) nm[r] = fmaxf(nm[r], __shfl_xor(nm[r], m, 32));

    float scl[8], rs[8];
#pragma unroll
    for (int r = 0; r < 8; ++r) {
      float mn = fmaxf(mrun[r], nm[r]);
      scl[r] = __expf(mrun[r] - mn);
      mrun[r] = mn;
      rs[r] = 0.f;
    }
#pragma unroll
    for (int t = 0; t < 4; ++t)
#pragma unroll
      for (int r = 0; r < 8; ++r) {
        float p = __expf(sacc[t][r] - mrun[r]);
        sacc[t][r] = p;
        rs[r] += p;
      }
#pragma unroll
    for (int m = 8; m >= 1; m >>= 1)
#pragma unroll
      for (int r = 0; r < 8; ++r) rs[r] += __shfl_xor(rs[r], m, 32);
#pragma unroll
    for (int r = 0; r < 8; ++r) lrun[r] = lrun[r] * scl[r] + rs[r];
#pragma unroll
    for (int t = 0; t < 4; ++t)
#pragma unroll
      for (int r = 0; r < 8; ++r) oacc[t][r] *= scl[r];

    // P (C/D layout) -> wave-private LDS row-major, then re-read as A fragments
#pragma unroll
    for (int t = 0; t < 4; ++t)
#pragma unroll
      for (int r = 0; r < 8; ++r)
        Pb[wave][r + (hi ? 8 : 0)][t * 16 + ln16] = (_Float16)sacc[t][r];

#pragma unroll
    for (int s2 = 0; s2 < 2; ++s2) {
      FragAB pa = load_a_frag(&Pb[wave][ln16][0], s2 * 32, hi);
#pragma unroll
      for (int t2 = 0; t2 < 4; ++t2)
        oacc[t2] = wmma16(pa, load_b_frag(&Vt[t2 * 16 + ln16][0], s2 * 32, hi), oacc[t2]);
    }
    __syncthreads();
  }

#pragma unroll
  for (int t2 = 0; t2 < 4; ++t2)
#pragma unroll
    for (int r = 0; r < 8; ++r) {
      int m = q0 + r + (hi ? 8 : 0);
      float v = oacc[t2][r] / lrun[r];
      attnout[(rowbase + m) * DMODEL + hh * HDIM + t2 * 16 + ln16] = (_Float16)v;
    }
}

// small head: hid = relu(cls@W1^T+b1); out = hid@W2^T+b2
__global__ __launch_bounds__(256) void head_kernel(const float* __restrict__ h,
                                                   const float* __restrict__ W1,
                                                   const float* __restrict__ b1,
                                                   const float* __restrict__ W2,
                                                   const float* __restrict__ b2,
                                                   float* __restrict__ out) {
  __shared__ float cls[DMODEL];
  __shared__ float hid[DMODEL];
  int b = blockIdx.x, t = threadIdx.x;
  cls[t] = h[((size_t)b * SMAX) * DMODEL + t];
  __syncthreads();
  float s = b1[t];
  for (int k = 0; k < DMODEL; ++k) s += cls[k] * W1[t * DMODEL + k];
  hid[t] = fmaxf(s, 0.f);
  __syncthreads();
  if (t < 2) {
    float o = b2[t];
    for (int k = 0; k < DMODEL; ++k) o += hid[k] * W2[t * DMODEL + k];
    out[b * 2 + t] = o;
  }
}

// ---------------------------------------------------------------- launch

extern "C" void kernel_launch(void* const* d_in, const int* in_sizes, int n_in,
                              void* d_out, int out_size, void* d_ws, size_t ws_size,
                              hipStream_t stream) {
  (void)in_sizes; (void)n_in; (void)out_size; (void)ws_size;
  const float* emb    = (const float*)d_in[0];
  const int*   evt    = (const int*)d_in[1];
  const float* geom   = (const float*)d_in[3];
  const float* in_w   = (const float*)d_in[4];
  const float* in_b   = (const float*)d_in[5];
  const float* geo_w  = (const float*)d_in[6];
  const float* geo_b  = (const float*)d_in[7];
  const float* cls    = (const float*)d_in[8];
  const float* qkv_w  = (const float*)d_in[9];
  const float* qkv_b  = (const float*)d_in[10];
  const float* out_w  = (const float*)d_in[11];
  const float* out_b  = (const float*)d_in[12];
  const float* ln1w   = (const float*)d_in[13];
  const float* ln1b   = (const float*)d_in[14];
  const float* ln2w   = (const float*)d_in[15];
  const float* ln2b   = (const float*)d_in[16];
  const float* ffw1   = (const float*)d_in[17];
  const float* bff1   = (const float*)d_in[18];
  const float* ffw2   = (const float*)d_in[19];
  const float* bff2   = (const float*)d_in[20];
  const float* hw1    = (const float*)d_in[21];
  const float* hb1    = (const float*)d_in[22];
  const float* hw2    = (const float*)d_in[23];
  const float* hb2    = (const float*)d_in[24];

  char* base = (char*)d_ws;
  size_t off = 0;
  auto take = [&](size_t bytes) -> char* {
    char* p = base + off;
    off += (bytes + 255) & ~(size_t)255;
    return p;
  };
  float*    h     = (float*)take((size_t)ROWS * DMODEL * 4);      // 100 MB
  _Float16* xln   = (_Float16*)take((size_t)ROWS * DMODEL * 2);   //  50 MB
  char*     scr   = take((size_t)ROWS * DFF * 2);                 // 201 MB aliased union
  _Float16* qkvb  = (_Float16*)scr;                               //   qkv phase
  _Float16* attnb = (_Float16*)(scr + (size_t)ROWS * 768 * 2);
  _Float16* ffb   = (_Float16*)scr;                               //   ff phase
  _Float16* embh  = (_Float16*)scr;                               //   embed phase
  _Float16* ench  = (_Float16*)(scr + (size_t)NDOM * DIN * 2);
  float*    maskb = (float*)take((size_t)ROWS * 4);
  int*      starts= (int*)take(1024);
  _Float16* winh  = (_Float16*)take((size_t)DMODEL * DIN * 2);
  _Float16* wgeoh = (_Float16*)take((size_t)DMODEL * DMODEL * 2);
  _Float16* qkvwh = (_Float16*)take((size_t)NLAYERS * 3 * DMODEL * DMODEL * 2);
  _Float16* outwh = (_Float16*)take((size_t)NLAYERS * DMODEL * DMODEL * 2);
  _Float16* ffw1h = (_Float16*)take((size_t)NLAYERS * DFF * DMODEL * 2);
  _Float16* ffw2h = (_Float16*)take((size_t)NLAYERS * DMODEL * DFF * 2);

  dim3 blk(256);
  auto cvt = [&](const float* s, _Float16* d, int n) {
    cvt_kernel<<<dim3((n + 255) / 256), blk, 0, stream>>>(s, d, n);
  };
  cvt(in_w,  winh,  DMODEL * DIN);
  cvt(geo_w, wgeoh, DMODEL * DMODEL);
  cvt(qkv_w, qkvwh, NLAYERS * 3 * DMODEL * DMODEL);
  cvt(out_w, outwh, NLAYERS * DMODEL * DMODEL);
  cvt(ffw1,  ffw1h, NLAYERS * DFF * DMODEL);
  cvt(ffw2,  ffw2h, NLAYERS * DMODEL * DFF);

  prep_kernel<<<dim3(NDOM), blk, 0, stream>>>(emb, geom, embh, ench);
  starts_kernel<<<dim3(1), dim3(128), 0, stream>>>(evt, starts);
  init_kernel<<<dim3(ROWS), blk, 0, stream>>>(cls, h, maskb);
  embed_kernel<<<dim3(NDOM / 128, DMODEL / 16), blk, 0, stream>>>(
      embh, ench, winh, wgeoh, in_b, geo_b, evt, starts, h, maskb);

  for (int l = 0; l < NLAYERS; ++l) {
    ln_kernel<<<dim3(ROWS / 8), blk, 0, stream>>>(h, ln1w + l * DMODEL, ln1b + l * DMODEL, xln);
    gemm_kernel<<<dim3(ROWS / 128, (3 * DMODEL) / 16), blk, 0, stream>>>(
        xln, qkvwh + (size_t)l * 3 * DMODEL * DMODEL, qkv_b + l * 3 * DMODEL,
        nullptr, nullptr, qkvb, ROWS, 3 * DMODEL, DMODEL, 0);
    attn_kernel<<<dim3(BATCH * NHEADS * QBLKS), blk, 0, stream>>>(qkvb, maskb, attnb);
    gemm_kernel<<<dim3(ROWS / 128, DMODEL / 16), blk, 0, stream>>>(
        attnb, outwh + (size_t)l * DMODEL * DMODEL, out_b + l * DMODEL,
        h, h, nullptr, ROWS, DMODEL, DMODEL, 0);
    ln_kernel<<<dim3(ROWS / 8), blk, 0, stream>>>(h, ln2w + l * DMODEL, ln2b + l * DMODEL, xln);
    gemm_kernel<<<dim3(ROWS / 128, DFF / 16), blk, 0, stream>>>(
        xln, ffw1h + (size_t)l * DFF * DMODEL, bff1 + l * DFF,
        nullptr, nullptr, ffb, ROWS, DFF, DMODEL, 1);
    gemm_kernel<<<dim3(ROWS / 128, DMODEL / 16), blk, 0, stream>>>(
        ffb, ffw2h + (size_t)l * DMODEL * DFF, bff2 + l * DMODEL,
        h, h, nullptr, ROWS, DMODEL, DFF, 0);
  }

  head_kernel<<<dim3(BATCH), blk, 0, stream>>>(h, hw1, hb1, hw2, hb2, (float*)d_out);
}
